// GAT_62156766708235
// MI455X (gfx1250) — matrix operand order repaired and checked by reading, not compiled, not run on hardware
//
#include <hip/hip_runtime.h>
#include <hip/hip_bf16.h>
#include <math.h>

// ---------------------------------------------------------------------------
// GAT forward on MI455X (gfx1250, wave32).
//  - Dense GEMMs run v_wmma_f32_16x16x32_f16 with f32 accumulate.
//    Operands are pre-staged in f16 in fragment-friendly layouts:
//      Ah [M, Kp]  (row-major, K padded to 32)  -> A-frag = 2x b128 loads
//      Bt [Np, Kp] (B transposed, N padded 16)  -> B-frag = 2x b128 loads
//    Inner loop: 4x global_load_b128 + 1x v_wmma, branch-free (EXEC all-1s).
//  - Edge softmax: 3 passes with native L2 atomics:
//      (1) segment-max via ordered-uint global_atomic_max_u32
//      (2) exp + segment-sum via global_atomic_add_f32 (p cached per edge)
//      (3) weighted scatter-add of messages via global_atomic_add_f32
//  - Whole working set (~170MB) is resident in the 192MB L2; HBM untouched
//    after first pass, so L2 atomic throughput is the roofline.
// ---------------------------------------------------------------------------

typedef __attribute__((ext_vector_type(16))) _Float16 v16h;
typedef __attribute__((ext_vector_type(8)))  _Float16 v8h;
typedef __attribute__((ext_vector_type(8)))  float    v8f;

#define NNODES  50000
#define NEDGES  800000
#define NGRAPHS 256
#define NEGSLP  0.2f
#define ORD_NEG_INF 0x007FFFFFu   // f2ord(-inf)

__device__ __forceinline__ unsigned f2ord(float f) {
    unsigned u = __float_as_uint(f);
    return (u & 0x80000000u) ? ~u : (u | 0x80000000u);
}
__device__ __forceinline__ float ord2f(unsigned v) {
    return __uint_as_float((v & 0x80000000u) ? (v ^ 0x80000000u) : ~v);
}
__device__ __forceinline__ float leaky(float a) {
    return a > 0.0f ? a : NEGSLP * a;
}

// ---------------------------------------------------------------------------
// WMMA GEMM:  out = act(Ah @ Bt^T + bias)
//   Ah: [M, Kp] f16 row-major (zero-padded K), M % 16 == 0
//   Bt: [Np, Kp] f16 (original B transposed; Np = ceil16(N), zero-padded)
//   C : optional f32 [M, N] output;  Ch: optional f16 [M, N] output
// Block = 128 threads = 4 waves; wave w owns C-tile (blockIdx.x, blockIdx.y*4+w).
// Fragment layouts per CDNA5 ISA 7.12.2 (lane = hi*16+lp):
//   A elem e -> A[tm*16+lp][kb + (e<8?0:16) + hi*8 + (e&7)]  = 2 contig 8-half runs
//   B elem e -> B[kb + (e<8?0:16) + hi*8 + (e&7)][tn*16+lp] = 2 contig 8-half runs of Bt[col]
//   C elem v -> C[tm*16 + hi*8 + v][tn*16+lp]
// ---------------------------------------------------------------------------
template <int ACT>   // 0 = identity, 1 = relu
__global__ void gemm_wmma_kernel(const _Float16* __restrict__ Ah,
                                 const _Float16* __restrict__ Bt,
                                 const float* __restrict__ bias,
                                 float* __restrict__ C,
                                 _Float16* __restrict__ Ch,
                                 int M, int Kp, int N)
{
    const int wave = threadIdx.x >> 5;
    const int lane = threadIdx.x & 31;
    const int tm   = blockIdx.x;
    const int tn   = blockIdx.y * 4 + wave;
    if (tn * 16 >= N) return;                 // wave-uniform exit

    const int hi = lane >> 4;
    const int lp = lane & 15;
    const int arow = tm * 16 + lp;            // < M (M multiple of 16)
    const int bcol = tn * 16 + lp;            // < Np (Bt padded to 16)

    const _Float16* pa = Ah + (size_t)arow * Kp + hi * 8;
    const _Float16* pb = Bt + (size_t)bcol * Kp + hi * 8;

    v8f c = {};
    for (int kb = 0; kb < Kp; kb += 32) {
        union { v16h v; v8h h[2]; } a, b;
        a.h[0] = *(const v8h*)(pa);           // K = kb + hi*8 + 0..7
        a.h[1] = *(const v8h*)(pa + 16);      // K = kb + 16 + hi*8 + 0..7
        b.h[0] = *(const v8h*)(pb);           // K = kb + hi*8 + 0..7
        b.h[1] = *(const v8h*)(pb + 16);      // K = kb + 16 + hi*8 + 0..7
        __builtin_prefetch(pb + 32, 0, 3);    // global_prefetch_b8 on next B chunk
        c = __builtin_amdgcn_wmma_f32_16x16x32_f16(
                /*neg_a=*/false, a.v, /*neg_b=*/false, b.v,
                /*c_mod=*/(short)0, c, /*reuse_a=*/false, /*reuse_b=*/false);
        pa += 32;
        pb += 32;
    }

    if (bcol < N) {
        const float bb = bias ? bias[bcol] : 0.0f;
#pragma unroll
        for (int v = 0; v < 8; ++v) {
            int row = tm * 16 + hi * 8 + v;
            float o = c[v] + bb;
            if (ACT == 1) o = fmaxf(o, 0.0f);
            if (C)  C [(size_t)row * N + bcol] = o;
            if (Ch) Ch[(size_t)row * N + bcol] = (_Float16)o;
        }
    }
}

// f32 [M,K] -> f16 [M,Kp] with zero pad
__global__ void cvtA_f16_kernel(const float* __restrict__ A, _Float16* __restrict__ Ah,
                                int M, int K, int Kp)
{
    long long idx = (long long)blockIdx.x * blockDim.x + threadIdx.x;
    if (idx >= (long long)M * Kp) return;
    int m = (int)(idx / Kp);
    int k = (int)(idx % Kp);
    Ah[idx] = (k < K) ? (_Float16)A[(size_t)m * K + k] : (_Float16)0.0f;
}

// f32 B [K,N] -> f16 Bt [Np,Kp] transposed, zero pad
__global__ void cvtBt_f16_kernel(const float* __restrict__ B, _Float16* __restrict__ Bt,
                                 int K, int N, int Kp, int Np)
{
    long long idx = (long long)blockIdx.x * blockDim.x + threadIdx.x;
    if (idx >= (long long)Np * Kp) return;
    int n = (int)(idx / Kp);
    int k = (int)(idx % Kp);
    Bt[idx] = (k < K && n < N) ? (_Float16)B[(size_t)k * N + n] : (_Float16)0.0f;
}

// ---------------------------------------------------------------------------
// Per-node attention logits: s_src[n,h] = sum_c h[n,h,c] * a_src[h,c] (and dst)
// ---------------------------------------------------------------------------
template <int H, int CC>
__global__ void attn_score_kernel(const float* __restrict__ h,
                                  const float* __restrict__ a_src,
                                  const float* __restrict__ a_dst,
                                  float* __restrict__ ssrc,
                                  float* __restrict__ sdst, int Nn)
{
    long long idx = (long long)blockIdx.x * blockDim.x + threadIdx.x;
    if (idx >= (long long)Nn * H) return;
    int n  = (int)(idx / H);
    int hh = (int)(idx % H);
    const float* hp = h + (size_t)n * H * CC + (size_t)hh * CC;
    float s0 = 0.0f, s1 = 0.0f;
#pragma unroll 4
    for (int c = 0; c < CC; ++c) {
        float v = hp[c];
        s0 += v * a_src[hh * CC + c];
        s1 += v * a_dst[hh * CC + c];
    }
    ssrc[idx] = s0;
    sdst[idx] = s1;
}

__global__ void fill_u32_kernel(unsigned* __restrict__ p, unsigned v, long long n)
{
    long long i = (long long)blockIdx.x * blockDim.x + threadIdx.x;
    if (i < n) p[i] = v;
}

// Pass 1: segment max of leaky(s_src[src]+s_dst[dst]) per (dst, head)
template <int H>
__global__ void edge_max_kernel(const int* __restrict__ ei, int E, int Nn,
                                const float* __restrict__ ssrc,
                                const float* __restrict__ sdst,
                                unsigned* __restrict__ mord)
{
    long long idx = (long long)blockIdx.x * blockDim.x + threadIdx.x;
    if (idx >= (long long)(E + Nn) * H) return;
    int e  = (int)(idx / H);
    int hh = (int)(idx % H);
    int src = (e < E) ? ei[e]     : (e - E);   // self loop for e >= E
    int dst = (e < E) ? ei[E + e] : (e - E);
    float a = leaky(ssrc[src * H + hh] + sdst[dst * H + hh]);
    atomicMax(&mord[dst * H + hh], f2ord(a));
}

// Pass 2: p = exp(a - m[dst]); cache p per edge; segment-sum into denom
template <int H>
__global__ void edge_expsum_kernel(const int* __restrict__ ei, int E, int Nn,
                                   const float* __restrict__ ssrc,
                                   const float* __restrict__ sdst,
                                   const unsigned* __restrict__ mord,
                                   float* __restrict__ walpha,
                                   float* __restrict__ den)
{
    long long idx = (long long)blockIdx.x * blockDim.x + threadIdx.x;
    if (idx >= (long long)(E + Nn) * H) return;
    int e  = (int)(idx / H);
    int hh = (int)(idx % H);
    int src = (e < E) ? ei[e]     : (e - E);
    int dst = (e < E) ? ei[E + e] : (e - E);
    float a = leaky(ssrc[src * H + hh] + sdst[dst * H + hh]);
    float p = __expf(a - ord2f(mord[dst * H + hh]));
    walpha[idx] = p;
    atomicAdd(&den[dst * H + hh], p);
}

// Pass 3: acc[dst, h, c] += (p / (den[dst,h]+eps)) * hfeat[src, h, c]
template <int H, int CC>
__global__ void edge_aggregate_kernel(const int* __restrict__ ei, int E, int Nn,
                                      const float* __restrict__ hfeat,
                                      const float* __restrict__ walpha,
                                      const float* __restrict__ den,
                                      float* __restrict__ acc)
{
    const int HC = H * CC;
    long long idx = (long long)blockIdx.x * blockDim.x + threadIdx.x;
    if (idx >= (long long)(E + Nn) * HC) return;
    int e = (int)(idx / HC);
    int j = (int)(idx % HC);
    int hh = j / CC;
    int src = (e < E) ? ei[e]     : (e - E);
    int dst = (e < E) ? ei[E + e] : (e - E);
    float w = walpha[(long long)e * H + hh] / (den[dst * H + hh] + 1e-16f);
    atomicAdd(&acc[(size_t)dst * HC + j], w * hfeat[(size_t)src * HC + j]);
}

// acc = elu(acc + bias); optional f16 copy for the next GEMM's A operand
__global__ void bias_elu_kernel(float* __restrict__ acc,
                                const float* __restrict__ bias,
                                _Float16* __restrict__ out16,
                                int Nn, int F)
{
    long long idx = (long long)blockIdx.x * blockDim.x + threadIdx.x;
    if (idx >= (long long)Nn * F) return;
    int f = (int)(idx % F);
    float v = acc[idx] + bias[f];
    v = v > 0.0f ? v : (__expf(v) - 1.0f);
    acc[idx] = v;
    if (out16) out16[idx] = (_Float16)v;
}

// global max pool per graph (ordered-uint atomic max)
__global__ void pool_max_kernel(const float* __restrict__ h,
                                const int* __restrict__ batch,
                                unsigned* __restrict__ pord, int Nn, int F)
{
    long long idx = (long long)blockIdx.x * blockDim.x + threadIdx.x;
    if (idx >= (long long)Nn * F) return;
    int n = (int)(idx / F);
    int f = (int)(idx % F);
    atomicMax(&pord[batch[n] * F + f], f2ord(h[idx]));
}

// decode ordered uints straight to f16 (feeds the MLP's first WMMA GEMM)
__global__ void ord_decode_f16_kernel(const unsigned* __restrict__ pord,
                                      _Float16* __restrict__ p, long long n)
{
    long long i = (long long)blockIdx.x * blockDim.x + threadIdx.x;
    if (i < n) p[i] = (_Float16)ord2f(pord[i]);
}

// ---------------------------------------------------------------------------
// Host-side orchestration
// ---------------------------------------------------------------------------
static inline unsigned blocks_for(long long n, int bs) {
    return (unsigned)((n + bs - 1) / bs);
}

extern "C" void kernel_launch(void* const* d_in, const int* in_sizes, int n_in,
                              void* d_out, int out_size, void* d_ws, size_t ws_size,
                              hipStream_t stream)
{
    const float* x     = (const float*)d_in[0];
    const int*   ei    = (const int*)  d_in[1];
    const int*   batch = (const int*)  d_in[2];
    const float* W1    = (const float*)d_in[3];
    const float* as1   = (const float*)d_in[4];
    const float* ad1   = (const float*)d_in[5];
    const float* b1    = (const float*)d_in[6];
    const float* W2    = (const float*)d_in[7];
    const float* as2   = (const float*)d_in[8];
    const float* ad2   = (const float*)d_in[9];
    const float* b2    = (const float*)d_in[10];
    const float* Wl1   = (const float*)d_in[11];
    const float* bl1   = (const float*)d_in[12];
    const float* Wl2   = (const float*)d_in[13];
    const float* bl2   = (const float*)d_in[14];
    const float* Wl3   = (const float*)d_in[15];
    const float* bl3   = (const float*)d_in[16];
    float* out = (float*)d_out;

    const int N = NNODES, E = NEDGES, G = NGRAPHS;
    const int ET = E + N;                     // edges + self loops

    // bump allocator over workspace
    char* wsp = (char*)d_ws;
    auto carve = [&](size_t bytes) -> void* {
        void* p = (void*)wsp;
        wsp += (bytes + 255) & ~(size_t)255;
        return p;
    };
    // f32 state
    float*    h1    = (float*)   carve((size_t)N * 64 * 4);
    float*    ss1   = (float*)   carve((size_t)N * 4 * 4);
    float*    sd1   = (float*)   carve((size_t)N * 4 * 4);
    unsigned* m1    = (unsigned*)carve((size_t)N * 4 * 4);
    float*    den1  = (float*)   carve((size_t)N * 4 * 4);
    float*    wal1  = (float*)   carve((size_t)ET * 4 * 4);
    float*    acc1  = (float*)   carve((size_t)N * 64 * 4);
    float*    h2    = (float*)   carve((size_t)N * 256 * 4);
    float*    ss2   = (float*)   carve((size_t)N * 4);
    float*    sd2   = (float*)   carve((size_t)N * 4);
    unsigned* m2    = (unsigned*)carve((size_t)N * 4);
    float*    den2  = (float*)   carve((size_t)N * 4);
    float*    wal2  = (float*)   carve((size_t)ET * 4);
    float*    acc2  = (float*)   carve((size_t)N * 256 * 4);
    unsigned* pord  = (unsigned*)carve((size_t)G * 256 * 4);
    // f16 staged GEMM operands
    _Float16* xh    = (_Float16*)carve((size_t)N * 32 * 2);      // x, Kp=32
    _Float16* W1t   = (_Float16*)carve((size_t)64 * 32 * 2);     // [Np=64, Kp=32]
    _Float16* h1h   = (_Float16*)carve((size_t)N * 64 * 2);      // elu layer1 out
    _Float16* W2t   = (_Float16*)carve((size_t)256 * 64 * 2);
    _Float16* poolh = (_Float16*)carve((size_t)G * 256 * 2);
    _Float16* Wl1t  = (_Float16*)carve((size_t)512 * 256 * 2);
    _Float16* z1h   = (_Float16*)carve((size_t)G * 512 * 2);
    _Float16* Wl2t  = (_Float16*)carve((size_t)1024 * 512 * 2);
    _Float16* z2h   = (_Float16*)carve((size_t)G * 1024 * 2);
    _Float16* Wl3t  = (_Float16*)carve((size_t)16 * 1024 * 2);   // Np=16, Kp=1024
    (void)ws_size; (void)n_in; (void)in_sizes; (void)out_size;

    const int BS = 256;
    const dim3 gemmB(128, 1, 1);

    // ---------------- stage f16 operands (tiny, L2-resident)
    cvtA_f16_kernel <<<blocks_for((long long)N * 32, BS), BS, 0, stream>>>(x,   xh,   N, 27, 32);
    cvtBt_f16_kernel<<<blocks_for((long long)64 * 32, BS), BS, 0, stream>>>(W1,  W1t,  27, 64, 32, 64);
    cvtBt_f16_kernel<<<blocks_for((long long)256 * 64, BS), BS, 0, stream>>>(W2,  W2t,  64, 256, 64, 256);
    cvtBt_f16_kernel<<<blocks_for((long long)512 * 256, BS), BS, 0, stream>>>(Wl1, Wl1t, 256, 512, 256, 512);
    cvtBt_f16_kernel<<<blocks_for((long long)1024 * 512, BS), BS, 0, stream>>>(Wl2, Wl2t, 512, 1024, 512, 1024);
    cvtBt_f16_kernel<<<blocks_for((long long)16 * 1024, BS), BS, 0, stream>>>(Wl3, Wl3t, 1024, 4, 1024, 16);

    // ---------------- GAT layer 1:  h1 = x @ W1  (27 -> 64, H=4, C=16)
    gemm_wmma_kernel<0><<<dim3(N / 16, 1), gemmB, 0, stream>>>(xh, W1t, nullptr, h1, nullptr, N, 32, 64);
    attn_score_kernel<4, 16><<<blocks_for((long long)N * 4, BS), BS, 0, stream>>>(h1, as1, ad1, ss1, sd1, N);
    fill_u32_kernel<<<blocks_for((long long)N * 4, BS), BS, 0, stream>>>(m1, ORD_NEG_INF, (long long)N * 4);
    fill_u32_kernel<<<blocks_for((long long)N * 4, BS), BS, 0, stream>>>((unsigned*)den1, 0u, (long long)N * 4);
    fill_u32_kernel<<<blocks_for((long long)N * 64, BS), BS, 0, stream>>>((unsigned*)acc1, 0u, (long long)N * 64);
    edge_max_kernel<4>   <<<blocks_for((long long)ET * 4, BS), BS, 0, stream>>>(ei, E, N, ss1, sd1, m1);
    edge_expsum_kernel<4><<<blocks_for((long long)ET * 4, BS), BS, 0, stream>>>(ei, E, N, ss1, sd1, m1, wal1, den1);
    edge_aggregate_kernel<4, 16><<<blocks_for((long long)ET * 64, BS), BS, 0, stream>>>(ei, E, N, h1, wal1, den1, acc1);
    bias_elu_kernel<<<blocks_for((long long)N * 64, BS), BS, 0, stream>>>(acc1, b1, h1h, N, 64);

    // ---------------- GAT layer 2:  h2 = elu_out @ W2  (64 -> 256, H=1)
    gemm_wmma_kernel<0><<<dim3(N / 16, 4), gemmB, 0, stream>>>(h1h, W2t, nullptr, h2, nullptr, N, 64, 256);
    attn_score_kernel<1, 256><<<blocks_for((long long)N, BS), BS, 0, stream>>>(h2, as2, ad2, ss2, sd2, N);
    fill_u32_kernel<<<blocks_for((long long)N, BS), BS, 0, stream>>>(m2, ORD_NEG_INF, (long long)N);
    fill_u32_kernel<<<blocks_for((long long)N, BS), BS, 0, stream>>>((unsigned*)den2, 0u, (long long)N);
    fill_u32_kernel<<<blocks_for((long long)N * 256, BS), BS, 0, stream>>>((unsigned*)acc2, 0u, (long long)N * 256);
    edge_max_kernel<1>   <<<blocks_for((long long)ET, BS), BS, 0, stream>>>(ei, E, N, ss2, sd2, m2);
    edge_expsum_kernel<1><<<blocks_for((long long)ET, BS), BS, 0, stream>>>(ei, E, N, ss2, sd2, m2, wal2, den2);
    edge_aggregate_kernel<1, 256><<<blocks_for((long long)ET * 256, BS), BS, 0, stream>>>(ei, E, N, h2, wal2, den2, acc2);
    bias_elu_kernel<<<blocks_for((long long)N * 256, BS), BS, 0, stream>>>(acc2, b2, nullptr, N, 256);

    // ---------------- global max pool per graph
    fill_u32_kernel<<<blocks_for((long long)G * 256, BS), BS, 0, stream>>>(pord, ORD_NEG_INF, (long long)G * 256);
    pool_max_kernel<<<blocks_for((long long)N * 256, BS), BS, 0, stream>>>(acc2, batch, pord, N, 256);
    ord_decode_f16_kernel<<<blocks_for((long long)G * 256, BS), BS, 0, stream>>>(pord, poolh, (long long)G * 256);

    // ---------------- MLP head (all WMMA, f16 chained through epilogues)
    gemm_wmma_kernel<1><<<dim3(G / 16,  8), gemmB, 0, stream>>>(poolh, Wl1t, bl1, nullptr, z1h, G, 256,  512);
    gemm_wmma_kernel<1><<<dim3(G / 16, 16), gemmB, 0, stream>>>(z1h,   Wl2t, bl2, nullptr, z2h, G, 512, 1024);
    gemm_wmma_kernel<0><<<dim3(G / 16,  1), gemmB, 0, stream>>>(z2h,   Wl3t, bl3, out, nullptr, G, 1024,   4);
}
